// MemoryAugmentedLSTM_32607391711289
// MI455X (gfx1250) — matrix operand order repaired and checked by reading, not compiled, not run on hardware
//
#include <hip/hip_runtime.h>
#include <stdint.h>

// ---------------------------------------------------------------------------
// MemoryAugmentedLSTM for MI455X (gfx1250): persistent bf16-WMMA kernel with
// Tensor-Data-Mover (TDM) staging of recurrent state into LDS.
// ---------------------------------------------------------------------------

typedef __attribute__((ext_vector_type(16))) __bf16 v16bf;
typedef __attribute__((ext_vector_type(8)))  float  v8f;
typedef __attribute__((ext_vector_type(4)))  unsigned v4u;
typedef __attribute__((ext_vector_type(8)))  int      v8i;
typedef __attribute__((ext_vector_type(4)))  int      v4i;

constexpr int T_SEQ = 1024;
constexpr int T_B   = 128;
constexpr int T_IN  = 512;
constexpr int T_H   = 512;
constexpr int T_4H  = 2048;
constexpr int T_OUT = 512;
constexpr int SLAB  = 528;    // LDS row stride (bf16): 512 data + 8-DWORD TDM pad
constexpr int NWG   = 32;     // persistent workgroups (8 M-tiles x 4 column groups)
constexpr int BLK   = 256;    // 8 waves of 32

#if __has_builtin(__builtin_amdgcn_tensor_load_to_lds)
#define HAVE_TDM 1
#else
#define HAVE_TDM 0
#endif

struct Params {
  const float* x; const float* stm; const float* ltm;
  const float* eb0; const float* eb1; const float* db0; const float* db1;
  const float* fb;  const float* pb;
  const __bf16 *Bx0, *Bh0, *Bx1, *Bh1;       // encoder weights, bf16, K x 2048 (K-major)
  const __bf16 *DBx0, *DBh0, *DBx1, *DBh1;   // decoder weights, bf16, K x 2048
  const __bf16 *Bfus, *Bproj;                // 1536x512 and 512x512, K-major
  __bf16 *h0a, *h0b, *h1a, *h1b;             // ping-pong hidden states, bf16 (128x512)
  __bf16 *mem, *d0, *d1;                     // fusion output / decoder hiddens (128x512)
  float* out;                                // 128 x 512 fp32
  unsigned* ctrl;                            // grid barrier state
};

// ---------------------------- device helpers -------------------------------

__device__ __forceinline__ float sigm(float x) { return 1.0f / (1.0f + __expf(-x)); }

__device__ __forceinline__ void grid_sync(unsigned* cnt, unsigned* gen) {
  __threadfence();
  __syncthreads();
  if (threadIdx.x == 0) {
    volatile unsigned* vgen = gen;
    unsigned g = *vgen;
    if (atomicAdd(cnt, 1u) == NWG - 1) {
      *cnt = 0u;
      __threadfence();
      atomicAdd(gen, 1u);
    } else {
      while (*vgen == g) { __builtin_amdgcn_s_sleep(8); }
    }
  }
  __syncthreads();
}

// --------- TDM: DMA a 16 x 512 bf16 tile (global, row stride 512) into an
// --------- LDS slab with 8-DWORD row padding (row stride = SLAB bf16 elems).
#if HAVE_TDM
__device__ __forceinline__ void tdm_load_tile16(const __bf16* gsrc, unsigned lds_byte_off) {
  unsigned long long ga = (unsigned long long)(size_t)gsrc;
  v4u g0;
  g0[0] = 1u;                                   // count=1, user descriptor, no gather
  g0[1] = lds_byte_off;                         // lds_addr (bytes)
  g0[2] = (unsigned)(ga & 0xFFFFFFFFu);         // global_addr[31:0]
  g0[3] = (unsigned)((ga >> 32) & 0x1FFFFFFu)   // global_addr[56:32]
          | (2u << 30);                         // type = 2 ("image")
  v8i g1;
  g1[0] = (int)((1u << 16)                      // data_size = 2 bytes (bf16)
              | (1u << 20)                      // pad_enable
              | (7u << 22)                      // pad_interval: 256 DWORDs (one row)
              | (7u << 25));                    // pad_amount: 8 DWORDs (16 bf16)
  g1[1] = (int)(512u << 16);                    // tensor_dim0 = 512 (bits 63:48)
  g1[2] = (int)(128u << 16);                    // tensor_dim0 hi=0 | tensor_dim1 = 128
  g1[3] = (int)(512u << 16);                    // tensor_dim1 hi=0 | tile_dim0 = 512
  g1[4] = (int)(16u);                           // tile_dim1 = 16, tile_dim2 = 0
  g1[5] = (int)(512u);                          // tensor_dim0_stride lo32 = 512 elems
  g1[6] = 0;                                    // stride hi | tensor_dim1_stride lo
  g1[7] = 0;
  v4i g2 = {0, 0, 0, 0};                        // 2D tile: groups 2/3 unused
  v4i g3 = {0, 0, 0, 0};
#if __has_include(<hip/amd_detail/amd_gfx1250_TDM.h>)
  v8i g4 = {0, 0, 0, 0, 0, 0, 0, 0};            // clang-23 / therock: 6-arg form
  __builtin_amdgcn_tensor_load_to_lds(g0, g1, g2, g3, g4, 0);
#else
  __builtin_amdgcn_tensor_load_to_lds(g0, g1, g2, g3, 0);  // ROCm 7.2: 5-arg form
#endif
}
#endif

// Cooperative fill of a 16 x 512 slab from fp32 global (with bf16 conversion)
__device__ __forceinline__ void fill_f32(__bf16* S, const float* src, int ld, int m0) {
  const int CH = 512 / 4;
  for (int i = threadIdx.x; i < 16 * CH; i += BLK) {
    int r = i / CH;
    int c = (i - r * CH) * 4;
    const float4 v = *(const float4*)(src + (size_t)(m0 + r) * ld + c);
    __bf16* dst = S + r * SLAB + c;
    dst[0] = (__bf16)v.x; dst[1] = (__bf16)v.y;
    dst[2] = (__bf16)v.z; dst[3] = (__bf16)v.w;
  }
}

// Fallback cooperative fill of a 16 x 512 slab from bf16 global
__device__ __forceinline__ void fill_bf16(__bf16* S, const __bf16* src, int m0) {
  const int CH = 512 / 8;
  for (int i = threadIdx.x; i < 16 * CH; i += BLK) {
    int r = i / CH;
    int c = (i - r * CH) * 8;
    *(uint4*)(S + r * SLAB + c) = *(const uint4*)(src + (size_t)(m0 + r) * T_H + c);
  }
}

// Stage a 16 x 512 bf16 tile into a slab: TDM if available, else VALU copy.
__device__ __forceinline__ void stage_h(__bf16* S, unsigned lds_off,
                                        const __bf16* src, int m0, int wid) {
#if HAVE_TDM
  (void)S;
  if (wid == 0) tdm_load_tile16(src + (size_t)m0 * T_H, lds_off);
#else
  (void)lds_off; (void)wid;
  fill_bf16(S, src, m0);
#endif
}

__device__ __forceinline__ void stage_commit(int wid) {
#if HAVE_TDM
  if (wid == 0) __builtin_amdgcn_s_wait_tensorcnt(0);
#else
  (void)wid;
#endif
  __syncthreads();
}

// A fragment (16x32 bf16) from an LDS slab
__device__ __forceinline__ v16bf ldsA(const __bf16* S, int k0, int lane) {
  const int row = lane & 15;
  const int kk  = k0 + ((lane >> 4) << 4);
  return *(const v16bf*)(S + row * SLAB + kk);
}

// 4-gate GEMM over K=512: acc[g] += A(slab) * Bm[:, g*512+n0 : +16]
__device__ __forceinline__ void gemm4(const __bf16* S, const __bf16* __restrict__ Bm,
                                      int n0, int lane, v8f* acc) {
#pragma unroll 4
  for (int k0 = 0; k0 < 512; k0 += 32) {
    v16bf a = ldsA(S, k0, lane);
#pragma unroll
    for (int g = 0; g < 4; ++g) {
      v16bf b;
#pragma unroll
      for (int s = 0; s < 16; ++s)
        b[s] = Bm[(size_t)(k0 + ((lane >> 4) << 4) + s) * T_4H + g * T_H + n0 + (lane & 15)];
      acc[g] = __builtin_amdgcn_wmma_f32_16x16x32_bf16(
          false, a, false, b, (short)0, acc[g], false, false);
    }
  }
}

// Single-tile GEMM over K=512 (ldn = 512) for fusion / projection
__device__ __forceinline__ v8f gemm1(const __bf16* S, const __bf16* __restrict__ Bm,
                                     int n0, int lane, v8f acc) {
#pragma unroll 4
  for (int k0 = 0; k0 < 512; k0 += 32) {
    v16bf a = ldsA(S, k0, lane);
    v16bf b;
#pragma unroll
    for (int s = 0; s < 16; ++s)
      b[s] = Bm[(size_t)(k0 + ((lane >> 4) << 4) + s) * T_H + n0 + (lane & 15)];
    acc = __builtin_amdgcn_wmma_f32_16x16x32_bf16(
        false, a, false, b, (short)0, acc, false, false);
  }
  return acc;
}

// LSTM nonlinearity on the wave's 16x16 tile; c kept in VGPRs, h written bf16
__device__ __forceinline__ void lstm_cell(const v8f* acc, const float* b, int n0,
                                          int lane, v8f& c, __bf16* hout, int m0) {
  const int col = n0 + (lane & 15);
  const float bi = b[col], bff = b[512 + col], bg = b[1024 + col], bo = b[1536 + col];
  const int rbase = m0 + ((lane >> 4) << 3);
#pragma unroll
  for (int e = 0; e < 8; ++e) {
    float iv = sigm(acc[0][e] + bi);
    float fv = sigm(acc[1][e] + bff);
    float gv = tanhf(acc[2][e] + bg);
    float ov = sigm(acc[3][e] + bo);
    float cn = fv * c[e] + iv * gv;
    c[e] = cn;
    hout[(size_t)(rbase + e) * T_H + col] = (__bf16)(ov * tanhf(cn));
  }
}

__device__ __forceinline__ void store_bias_bf16(v8f a, const float* b, int n0,
                                                int lane, __bf16* out, int m0) {
  const int col = n0 + (lane & 15);
  const float bb = b[col];
  const int rbase = m0 + ((lane >> 4) << 3);
#pragma unroll
  for (int e = 0; e < 8; ++e)
    out[(size_t)(rbase + e) * T_H + col] = (__bf16)(a[e] + bb);
}

__device__ __forceinline__ void store_bias_f32(v8f a, const float* b, int n0,
                                               int lane, float* out, int m0) {
  const int col = n0 + (lane & 15);
  const float bb = b[col];
  const int rbase = m0 + ((lane >> 4) << 3);
#pragma unroll
  for (int e = 0; e < 8; ++e)
    out[(size_t)(rbase + e) * T_OUT + col] = a[e] + bb;
}

// ------------------------------- kernels -----------------------------------

// W (N x K, fp32, row-major) -> Bm (K x N, bf16, K-major): Bm[k*N+n] = W[n*K+k]
__global__ void convert_transpose(const float* __restrict__ W,
                                  __bf16* __restrict__ Bm, int N, int K) {
  size_t total = (size_t)N * K;
  for (size_t i = blockIdx.x * (size_t)blockDim.x + threadIdx.x; i < total;
       i += (size_t)gridDim.x * blockDim.x) {
    size_t k = i / N;
    size_t n = i - k * N;
    Bm[i] = (__bf16)W[n * (size_t)K + k];
  }
}

__global__ void zero_init(unsigned* ctrl, unsigned* dst, int nwords) {
  int tid = blockIdx.x * blockDim.x + threadIdx.x;
  if (tid < 64) ctrl[tid] = 0u;
  for (int i = tid; i < nwords; i += gridDim.x * blockDim.x) dst[i] = 0u;
}

__global__ __launch_bounds__(BLK) void lstm_persistent(Params p) {
  // Three 16 x SLAB bf16 slabs; slab i sits at LDS byte offset i*16*SLAB*2
  // (single shared object => allocated at LDS offset 0, as the TDM D# needs).
  __shared__ __attribute__((aligned(64))) __bf16 SHM[3 * 16 * SLAB];
  __bf16* S0 = SHM;
  __bf16* S1 = SHM + 16 * SLAB;
  __bf16* S2 = SHM + 2 * 16 * SLAB;
  const unsigned OFF0 = 0u;
  const unsigned OFF1 = (unsigned)(16 * SLAB * 2);
  const unsigned OFF2 = (unsigned)(2 * 16 * SLAB * 2);

  const int lane = threadIdx.x & 31;
  const int wid  = threadIdx.x >> 5;
  const int mt   = blockIdx.x >> 2;          // 0..7  -> batch rows [mt*16, +16)
  const int cg   = blockIdx.x & 3;           // 0..3  -> hidden column group
  const int m0   = mt * 16;
  const int n0   = (cg * 8 + wid) * 16;      // this wave's hidden-column base

  unsigned* cnt = p.ctrl;
  unsigned* gen = p.ctrl + 16;

  const v8f zero = {};
  v8f c0 = zero, c1 = zero;                  // cell states live in VGPRs all run
  v8f acc[4];

  __bf16* h0buf[2] = {p.h0a, p.h0b};
  __bf16* h1buf[2] = {p.h1a, p.h1b};

#pragma unroll 1
  for (int t = 0; t < T_SEQ; ++t) {
    const __bf16* h0r = h0buf[(t + 1) & 1];
    __bf16*       h0w = h0buf[t & 1];
    const __bf16* h1r = h1buf[(t + 1) & 1];
    __bf16*       h1w = h1buf[t & 1];

    if (t + 1 < T_SEQ && lane == 0)          // warm L2/WGP$ for next timestep slab
      __builtin_prefetch(p.x + (size_t)(t + 1) * T_B * T_IN + (size_t)(m0 + wid * 2) * T_IN, 0, 1);

    // ---------------- encoder layer 0 ----------------
    stage_h(S1, OFF1, h0r, m0, wid);                    // TDM DMA overlaps VALU fill
    fill_f32(S0, p.x + (size_t)t * T_B * T_IN, T_IN, m0);
    stage_commit(wid);
    acc[0] = zero; acc[1] = zero; acc[2] = zero; acc[3] = zero;
    gemm4(S0, p.Bx0, n0, lane, acc);
    gemm4(S1, p.Bh0, n0, lane, acc);
    lstm_cell(acc, p.eb0, n0, lane, c0, h0w, m0);
    grid_sync(cnt, gen);

    // ---------------- encoder layer 1 ----------------
    stage_h(S0, OFF0, h0w, m0, wid);
    stage_h(S1, OFF1, h1r, m0, wid);
    stage_commit(wid);
    acc[0] = zero; acc[1] = zero; acc[2] = zero; acc[3] = zero;
    gemm4(S0, p.Bx1, n0, lane, acc);
    gemm4(S1, p.Bh1, n0, lane, acc);
    lstm_cell(acc, p.eb1, n0, lane, c1, h1w, m0);
    grid_sync(cnt, gen);
  }

  const __bf16* hf0 = h0buf[(T_SEQ - 1) & 1];
  const __bf16* hf1 = h1buf[(T_SEQ - 1) & 1];

  // ---------------- memory fusion: [h1 | stm | ltm] @ fus_W^T + fus_b --------
  stage_h(S0, OFF0, hf1, m0, wid);
  fill_f32(S1, p.stm, T_H, m0);
  fill_f32(S2, p.ltm, T_H, m0);
  stage_commit(wid);
  {
    v8f a = gemm1(S0, p.Bfus, n0, lane, zero);
    a = gemm1(S1, p.Bfus + (size_t)512 * T_H, n0, lane, a);
    a = gemm1(S2, p.Bfus + (size_t)1024 * T_H, n0, lane, a);
    store_bias_bf16(a, p.fb, n0, lane, p.mem, m0);
  }
  grid_sync(cnt, gen);

  // ---------------- decoder cell 0 (state = encoder h0,c0) ------------------
  stage_h(S0, OFF0, p.mem, m0, wid);
  stage_h(S1, OFF1, hf0, m0, wid);
  stage_commit(wid);
  acc[0] = zero; acc[1] = zero; acc[2] = zero; acc[3] = zero;
  gemm4(S0, p.DBx0, n0, lane, acc);
  gemm4(S1, p.DBh0, n0, lane, acc);
  lstm_cell(acc, p.db0, n0, lane, c0, p.d0, m0);
  grid_sync(cnt, gen);

  // ---------------- decoder cell 1 (state = encoder h1,c1) ------------------
  stage_h(S0, OFF0, p.d0, m0, wid);
  stage_h(S1, OFF1, hf1, m0, wid);
  stage_commit(wid);
  acc[0] = zero; acc[1] = zero; acc[2] = zero; acc[3] = zero;
  gemm4(S0, p.DBx1, n0, lane, acc);
  gemm4(S1, p.DBh1, n0, lane, acc);
  lstm_cell(acc, p.db1, n0, lane, c1, p.d1, m0);
  grid_sync(cnt, gen);

  // ---------------- projection: d1 @ proj_W^T + proj_b ----------------------
  stage_h(S0, OFF0, p.d1, m0, wid);
  stage_commit(wid);
  {
    v8f a = gemm1(S0, p.Bproj, n0, lane, zero);
    store_bias_f32(a, p.pb, n0, lane, p.out, m0);
  }
}

// ------------------------------ launcher -----------------------------------

extern "C" void kernel_launch(void* const* d_in, const int* in_sizes, int n_in,
                              void* d_out, int out_size, void* d_ws, size_t ws_size,
                              hipStream_t stream) {
  (void)in_sizes; (void)n_in; (void)out_size; (void)ws_size;

  char* ws = (char*)d_ws;
  size_t off = 0;
  auto alloc = [&](size_t bytes) -> char* {
    off = (off + 255) & ~(size_t)255;
    char* r = ws + off;
    off += bytes;
    return r;
  };

  constexpr size_t W4H = (size_t)512 * 2048 * sizeof(__bf16);   // 2 MB each
  constexpr size_t HST = (size_t)T_B * T_H * sizeof(__bf16);    // 128 KB each

  unsigned* ctrl  = (unsigned*)alloc(256);
  __bf16* Bx0   = (__bf16*)alloc(W4H);
  __bf16* Bh0   = (__bf16*)alloc(W4H);
  __bf16* Bx1   = (__bf16*)alloc(W4H);
  __bf16* Bh1   = (__bf16*)alloc(W4H);
  __bf16* DBx0  = (__bf16*)alloc(W4H);
  __bf16* DBh0  = (__bf16*)alloc(W4H);
  __bf16* DBx1  = (__bf16*)alloc(W4H);
  __bf16* DBh1  = (__bf16*)alloc(W4H);
  __bf16* Bfus  = (__bf16*)alloc((size_t)1536 * 512 * sizeof(__bf16));
  __bf16* Bproj = (__bf16*)alloc((size_t)512 * 512 * sizeof(__bf16));
  __bf16* hblk  = (__bf16*)alloc(4 * HST);   // h0a,h0b,h1a,h1b contiguous (zeroed)
  __bf16* mem   = (__bf16*)alloc(HST);
  __bf16* d0b   = (__bf16*)alloc(HST);
  __bf16* d1b   = (__bf16*)alloc(HST);

  __bf16* h0a = hblk;
  __bf16* h0b = hblk + T_B * T_H;
  __bf16* h1a = hblk + 2 * T_B * T_H;
  __bf16* h1b = hblk + 3 * T_B * T_H;

  // --- weight transpose + fp32 -> bf16 conversion (L2-resident afterwards) ---
  dim3 cb(256), cg(512);
  convert_transpose<<<cg, cb, 0, stream>>>((const float*)d_in[3],  Bx0,   2048, 512);
  convert_transpose<<<cg, cb, 0, stream>>>((const float*)d_in[4],  Bh0,   2048, 512);
  convert_transpose<<<cg, cb, 0, stream>>>((const float*)d_in[6],  Bx1,   2048, 512);
  convert_transpose<<<cg, cb, 0, stream>>>((const float*)d_in[7],  Bh1,   2048, 512);
  convert_transpose<<<cg, cb, 0, stream>>>((const float*)d_in[9],  DBx0,  2048, 512);
  convert_transpose<<<cg, cb, 0, stream>>>((const float*)d_in[10], DBh0,  2048, 512);
  convert_transpose<<<cg, cb, 0, stream>>>((const float*)d_in[12], DBx1,  2048, 512);
  convert_transpose<<<cg, cb, 0, stream>>>((const float*)d_in[13], DBh1,  2048, 512);
  convert_transpose<<<cg, cb, 0, stream>>>((const float*)d_in[15], Bfus,   512, 1536);
  convert_transpose<<<cg, cb, 0, stream>>>((const float*)d_in[17], Bproj,  512, 512);

  zero_init<<<128, 256, 0, stream>>>(ctrl, (unsigned*)hblk, (int)(4 * HST / 4));

  Params p;
  p.x    = (const float*)d_in[0];
  p.stm  = (const float*)d_in[1];
  p.ltm  = (const float*)d_in[2];
  p.eb0  = (const float*)d_in[5];
  p.eb1  = (const float*)d_in[8];
  p.db0  = (const float*)d_in[11];
  p.db1  = (const float*)d_in[14];
  p.fb   = (const float*)d_in[16];
  p.pb   = (const float*)d_in[18];
  p.Bx0 = Bx0; p.Bh0 = Bh0; p.Bx1 = Bx1; p.Bh1 = Bh1;
  p.DBx0 = DBx0; p.DBh0 = DBh0; p.DBx1 = DBx1; p.DBh1 = DBh1;
  p.Bfus = Bfus; p.Bproj = Bproj;
  p.h0a = h0a; p.h0b = h0b; p.h1a = h1a; p.h1b = h1b;
  p.mem = mem; p.d0 = d0b; p.d1 = d1b;
  p.out = (float*)d_out;
  p.ctrl = ctrl;

  lstm_persistent<<<NWG, BLK, 0, stream>>>(p);
}
